// PaLaDeformConv2d_48936857370777
// MI455X (gfx1250) — matrix operand order repaired and checked by reading, not compiled, not run on hardware
//
#include <hip/hip_runtime.h>
#include <math.h>

typedef unsigned short u16;
typedef __attribute__((ext_vector_type(8)))  __bf16 bf16x8;
typedef __attribute__((ext_vector_type(16))) __bf16 bf16x16;
typedef __attribute__((ext_vector_type(8)))  float  f32x8;

// ---------------- problem constants ----------------
static constexpr int Cc   = 64;      // input channels
static constexpr int Oc   = 64;      // output channels
static constexpr int KKc  = 9;       // 3x3 kernel positions
static constexpr int Hc   = 160;
static constexpr int Wc   = 160;
static constexpr int HW   = Hc * Wc;         // 25600
static constexpr int Nmat = 256;             // 128 (w_m) + 128 (w_n) GEMM columns
static constexpr int Ktot = Cc * KKc;        // 576
static constexpr int MT   = 32;              // pixels per workgroup (M tile)
static constexpr int ASTR = Ktot + 8;        // padded A row stride (bf16), 584*2B = 16B-aligned
static constexpr int BSTR = 40;              // padded B row stride per 32-K chunk (80B, 16B-aligned)
static constexpr int RSTR = Nmat + 1;        // f32 result stride (bank-friendly)

__device__ __forceinline__ u16 f2bf(float f) {
    unsigned u = __float_as_uint(f);
    unsigned r = u + 0x7FFFu + ((u >> 16) & 1u);   // round-to-nearest-even
    return (u16)(r >> 16);
}

__device__ __forceinline__ bf16x16 cat8(bf16x8 lo, bf16x8 hi) {
    return __builtin_shufflevector(lo, hi, 0,1,2,3,4,5,6,7,8,9,10,11,12,13,14,15);
}

// ---------------- weight pack: fp32 [128][C][3][3] x2 -> bf16 [256][kk*64+c] ----------------
__global__ void pala_pack_weights(const float* __restrict__ w_m,
                                  const float* __restrict__ w_n,
                                  u16* __restrict__ Wp) {
    int n = blockIdx.x;            // 0..255
    int t = threadIdx.x;           // 0..575 == c*9+kk
    const float* src = (n < 128) ? (w_m + (size_t)n * Ktot)
                                 : (w_n + (size_t)(n - 128) * Ktot);
    int c = t / KKc, kk = t % KKc;
    Wp[(size_t)n * Ktot + kk * Cc + c] = f2bf(src[t]);
}

// ---------------- fused: offsets -> bilinear sample -> WMMA GEMM -> Pade epilogue ----------------
__global__ __launch_bounds__(256)
void pala_fused(const float* __restrict__ x,
                const float* __restrict__ off_w,
                const float* __restrict__ off_b,
                const float* __restrict__ w0,
                const u16*   __restrict__ Wp,
                float* __restrict__ out) {
    __shared__ __align__(16) unsigned char smem[MT * ASTR * 2      // A tile  (37376 B)
                                                + Nmat * BSTR * 2  // B chunk (20480 B)
                                                + MT * KKc * 16];  // meta    ( 4608 B)
    u16*    Alds = (u16*)smem;
    u16*    Blds = (u16*)(smem + MT * ASTR * 2);
    float4* meta = (float4*)(smem + MT * ASTR * 2 + Nmat * BSTR * 2);
    float*  Res  = (float*)smem;                                   // reused after GEMM

    const int tid = threadIdx.x;
    const int wg  = blockIdx.x;

    // ---- phase 1: 1x1 offset conv (8 lanes per pixel) + clamp + tanh soft-limit ----
    {
        int pix = tid >> 3, l8 = tid & 7;
        int p = wg * MT + pix;
        int b = p / HW, hw = p % HW, h = hw / Wc, w = hw % Wc;
        const float* xb = x + (size_t)b * Cc * HW + (size_t)h * Wc + w;
        float acc[18];
        #pragma unroll
        for (int o = 0; o < 18; ++o) acc[o] = 0.f;
        #pragma unroll
        for (int i = 0; i < 8; ++i) {
            int c = l8 * 8 + i;
            float xv = xb[(size_t)c * HW];
            #pragma unroll
            for (int o = 0; o < 18; ++o)
                acc[o] = fmaf(xv, off_w[o * Cc + c], acc[o]);
        }
        #pragma unroll
        for (int o = 0; o < 18; ++o) {
            acc[o] += __shfl_xor(acc[o], 1, 8);
            acc[o] += __shfl_xor(acc[o], 2, 8);
            acc[o] += __shfl_xor(acc[o], 4, 8);
        }
        float fh = (float)h, fw = (float)w;
        for (int kk = l8; kk < KKc; kk += 8) {
            float dy = acc[2 * kk]     + off_b[2 * kk];
            float dx = acc[2 * kk + 1] + off_b[2 * kk + 1];
            dy = fminf(fmaxf(dy, -fh), (float)Hc - fh);
            dx = fminf(fmaxf(dx, -fw), (float)Wc - fw);
            if (fabsf(dy) >= 8.f) dy = 8.f * tanhf(dy * 0.125f);
            if (fabsf(dx) >= 8.f) dx = 8.f * tanhf(dx * 0.125f);
            float py = fh - 1.f + (float)(kk / 3) + dy;
            float px = fw - 1.f + (float)(kk % 3) + dx;
            float y0 = floorf(py), x0 = floorf(px);
            meta[pix * KKc + kk] = make_float4(y0, x0, py - y0, px - x0);
        }
    }
    __syncthreads();

    // ---- phase 2: bilinear sample -> bf16 A tile in LDS (2 channels/thread, packed b32 stores) ----
    {
        int cpair = tid & 31, pg = tid >> 5;
        int c0 = cpair * 2;
        unsigned* A32 = (unsigned*)Alds;
        for (int pix = pg; pix < MT; pix += 8) {
            int p = wg * MT + pix;
            int b = p / HW;
            const float* x0p = x + (size_t)(b * Cc + c0) * HW;
            const float* x1p = x0p + HW;
            #pragma unroll
            for (int kk = 0; kk < KKc; ++kk) {
                float4 m = meta[pix * KKc + kk];
                int   y0 = (int)m.x, xx0 = (int)m.y;
                float wy = m.z, wx = m.w;
                float cw[4] = { (1.f - wy) * (1.f - wx), (1.f - wy) * wx,
                                wy * (1.f - wx),         wy * wx };
                float s0 = 0.f, s1 = 0.f;
                #pragma unroll
                for (int cor = 0; cor < 4; ++cor) {
                    int yy = y0 + (cor >> 1);
                    int xx = xx0 + (cor & 1);
                    if ((unsigned)yy < (unsigned)Hc && (unsigned)xx < (unsigned)Wc) {
                        int idx = yy * Wc + xx;
                        s0 = fmaf(cw[cor], x0p[idx], s0);
                        s1 = fmaf(cw[cor], x1p[idx], s1);
                    }
                }
                unsigned packed = (unsigned)f2bf(s0) | ((unsigned)f2bf(s1) << 16);
                A32[pix * (ASTR / 2) + kk * 32 + cpair] = packed;
            }
        }
    }

    // ---- phase 3: GEMM  M=32 x N=256 x K=576 via v_wmma_f32_16x16x32_bf16 ----
    const int wv = tid >> 5, lane = tid & 31;
    const int lhalf = lane >> 4, l15 = lane & 15;

    f32x8 acc[2][2];
    #pragma unroll
    for (int mi = 0; mi < 2; ++mi)
        #pragma unroll
        for (int j = 0; j < 2; ++j)
            #pragma unroll
            for (int r = 0; r < 8; ++r) acc[mi][j][r] = 0.f;

    for (int ks = 0; ks < Ktot / 32; ++ks) {
        __syncthreads();  // A ready (iter 0) / previous Blds reads done
        {   // cooperative B chunk load: row n = tid, 32 bf16 = 4x b128
            const uint4* src = (const uint4*)(Wp + (size_t)tid * Ktot + ks * 32);
            uint4* dst = (uint4*)(Blds + tid * BSTR);
            dst[0] = src[0]; dst[1] = src[1]; dst[2] = src[2]; dst[3] = src[3];
        }
        __syncthreads();

        bf16x16 af[2], bf[2];
        #pragma unroll
        for (int mi = 0; mi < 2; ++mi) {
            // A 16x32 bf16 layout: lanes<16 hold K0+[0..8)+[16..24) of row l15; lanes>=16 the other halves
            const u16* ap = Alds + (mi * 16 + l15) * ASTR + ks * 32 + lhalf * 8;
            bf16x8 lo = *(const bf16x8*)ap;
            bf16x8 hi = *(const bf16x8*)(ap + 16);
            af[mi] = cat8(lo, hi);
        }
        #pragma unroll
        for (int j = 0; j < 2; ++j) {
            // B 32x16 bf16 layout: lanes<16 hold K0+[0..8)+[16..24) of column l15; lanes>=16 the other halves
            const u16* bp = Blds + ((wv * 2 + j) * 16 + l15) * BSTR + lhalf * 8;
            bf16x8 lo = *(const bf16x8*)bp;
            bf16x8 hi = *(const bf16x8*)(bp + 16);
            bf[j] = cat8(lo, hi);
        }
        #pragma unroll
        for (int mi = 0; mi < 2; ++mi)
            #pragma unroll
            for (int j = 0; j < 2; ++j)
                acc[mi][j] = __builtin_amdgcn_wmma_f32_16x16x32_bf16(
                    false, af[mi], false, bf[j], (short)0, acc[mi][j], false, false);
    }

    // ---- phase 4: dump accumulators to LDS, apply after_poly_a + Pade division ----
    __syncthreads();
    #pragma unroll
    for (int mi = 0; mi < 2; ++mi)
        #pragma unroll
        for (int j = 0; j < 2; ++j)
            #pragma unroll
            for (int r = 0; r < 8; ++r) {
                int row = mi * 16 + r + lhalf * 8;        // C/D layout: lanes>=16 hold M=r+8
                int col = (wv * 2 + j) * 16 + l15;
                Res[row * RSTR + col] = acc[mi][j][r];
            }
    __syncthreads();
    {
        int pix = tid & 31, cg = tid >> 5;
        int p = wg * MT + pix;
        int b = p / HW, hw = p % HW, h = hw / Wc, w = hw % Wc;
        float* op = out + (size_t)b * Oc * HW + (size_t)h * Wc + w;
        const float* r = Res + pix * RSTR;
        for (int c = cg; c < Oc; c += 8) {
            float pm = r[c] + r[c + 64] + w0[c];
            float qn = fabsf(r[c + 128]) + fabsf(r[c + 192]);
            op[(size_t)c * HW] = pm / (1.f + qn);
        }
    }
}

extern "C" void kernel_launch(void* const* d_in, const int* in_sizes, int n_in,
                              void* d_out, int out_size, void* d_ws, size_t ws_size,
                              hipStream_t stream) {
    const float* x     = (const float*)d_in[0];
    const float* off_w = (const float*)d_in[1];
    const float* off_b = (const float*)d_in[2];
    const float* w_m   = (const float*)d_in[3];
    const float* w_n   = (const float*)d_in[4];
    const float* w0    = (const float*)d_in[5];
    float* out = (float*)d_out;
    u16*   Wp  = (u16*)d_ws;   // 256*576 bf16 = 294,912 B packed weights

    pala_pack_weights<<<dim3(Nmat), dim3(Ktot), 0, stream>>>(w_m, w_n, Wp);

    const int npix = 4 * HW;               // B*H*W = 102400, divisible by MT
    pala_fused<<<dim3(npix / MT), dim3(256), 0, stream>>>(x, off_w, off_b, w0, Wp, out);
}
